// LSTM_75780402970950
// MI455X (gfx1250) — compile-verified, matrix operand order checked by reading, not run on hardware
//
#include <hip/hip_runtime.h>
#include <stdint.h>

// Problem dims (match reference)
#define B_   32
#define T_   256
#define IN_  512
#define HID_ 1024
#define OUT_ 512
#define G4_  (4 * HID_)

#define CHUNK_ 256   // K-columns staged into LDS per buffer (16KB)
#define APAD_  8     // +16B row pad: rotates LDS banks by 4/row -> conflict-free b128

typedef __attribute__((ext_vector_type(16))) __bf16 v16bf;
typedef __attribute__((ext_vector_type(8)))  __bf16 v8bf;
typedef __attribute__((ext_vector_type(8)))  float  v8f;

union ABu { v16bf v; v8bf h[2]; };

__device__ __forceinline__ __bf16 to_bf16(float f) {
  union { float f; uint32_t u; } v; v.f = f;
  // round-to-nearest-even truncation fp32 -> bf16
  uint32_t r = v.u + 0x7FFFu + ((v.u >> 16) & 1u);
  union { uint16_t s; __bf16 b; } o; o.s = (uint16_t)(r >> 16);
  return o.b;
}

__device__ __forceinline__ float sigmoidf_(float x) {
  return 1.0f / (1.0f + __expf(-x));
}

// ---------------------------------------------------------------------------
// Async-DMA stage of one 32 x CHUNK_ bf16 A block into LDS.
// 128 threads x 8 x 16B = 16KB; thread t owns row (t>>2) and column chunk
// (t&3)*8 within each 32-column group. 8 issues per wave -> ASYNCcnt += 8.
// VDST is the 32-bit LDS byte address (low 32 bits of a flat pointer to
// __shared__); VADDR is the 64-bit global addr.
// ---------------------------------------------------------------------------
__device__ __forceinline__ void issue_stage(__bf16 (*sh)[CHUNK_ + APAD_],
                                            const __bf16* __restrict__ A,
                                            int lda, int k0, int tid) {
  const int row = tid >> 2;
  const int c8  = (tid & 3) * 8;
  const __bf16* grow = A + (size_t)row * lda + k0;
#pragma unroll
  for (int it = 0; it < CHUNK_ / 32; ++it) {
    const int col = it * 32 + c8;
    const uint32_t lds_off = (uint32_t)(uintptr_t)(&sh[row][col]);
    asm volatile("global_load_async_to_lds_b128 %0, %1, off"
                 :: "v"(lds_off), "v"(grow + col) : "memory");
  }
}

// ---------------------------------------------------------------------------
// Workgroup GEMM segment: every wave accumulates a 32x16 fp32 tile of
// A[32,K] * B^T (B row-major [N,K]) with V_WMMA_F32_16X16X32_BF16.
// A (shared by all 4 waves) is staged via double-buffered async DMA; the
// 8-iteration inner block is barrier-free so the compiler software-pipelines
// the unique per-wave weight stream (global_load_b128 + global_prefetch_b8)
// and ds_load_b128 fragments against the WMMAs.
// bf16 fragment layout per ISA: lanes 0-15 carry K chunks {0..7,16..23},
// lanes 16-31 carry {8..15,24..31} -> two contiguous 16B reads each.
// ---------------------------------------------------------------------------
__device__ __forceinline__ void wave_gemm_lds(v8f& acc0, v8f& acc1,
                                              const __bf16* __restrict__ A, int lda,
                                              const __bf16* __restrict__ Brow, int K,
                                              int lane, int tid,
                                              __bf16 (*shA)[32][CHUNK_ + APAD_]) {
  const int m   = lane & 15;
  const int klo = (lane < 16) ? 0 : 8;
  const int nstages = K / CHUNK_;   // constant-folds (K is constexpr at call sites)

  issue_stage(shA[0], A, lda, 0, tid);

  for (int s = 0; s < nstages; ++s) {
    const int buf = s & 1;
    if (s + 1 < nstages) {
      // Next stage into the buffer read in the previous stage (safe: end-of-stage
      // barrier below already passed for it). 8 newer issues in flight ->
      // waiting asynccnt<=8 proves the current stage's 8 (older) landed.
      issue_stage(shA[buf ^ 1], A, lda, (s + 1) * CHUNK_, tid);
      asm volatile("s_wait_asynccnt 0x8" ::: "memory");
    } else {
      asm volatile("s_wait_asynccnt 0x0" ::: "memory");  // full drain on last stage
    }
    __syncthreads();   // all 4 waves' DMA sub-copies of this buffer visible

    const __bf16* bbase = Brow + s * CHUNK_;
#pragma unroll
    for (int kk = 0; kk < CHUNK_; kk += 32) {
      ABu a0, a1, b;
      a0.h[0] = *(const v8bf*)&shA[buf][m][kk + klo];
      a0.h[1] = *(const v8bf*)&shA[buf][m][kk + klo + 16];
      a1.h[0] = *(const v8bf*)&shA[buf][m + 16][kk + klo];
      a1.h[1] = *(const v8bf*)&shA[buf][m + 16][kk + klo + 16];
      b.h[0]  = *(const v8bf*)(bbase + kk + klo);
      b.h[1]  = *(const v8bf*)(bbase + kk + klo + 16);
      __builtin_prefetch(bbase + kk + 256, 0, 1);   // weight stream L2 -> L0
      acc0 = __builtin_amdgcn_wmma_f32_16x16x32_bf16(false, a0.v, false, b.v,
                                                     (short)0, acc0, false, false);
      acc1 = __builtin_amdgcn_wmma_f32_16x16x32_bf16(false, a1.v, false, b.v,
                                                     (short)0, acc1, false, false);
    }
    __syncthreads();   // all waves done reading buf before DMA overwrites it
  }
}

// Direct-global variant (one-shot readout; staging is moot there).
__device__ __forceinline__ void wave_gemm(v8f& acc0, v8f& acc1,
                                          const __bf16* __restrict__ A, int lda,
                                          const __bf16* __restrict__ Brow, int K,
                                          int lane) {
  const int m   = lane & 15;
  const int klo = (lane < 16) ? 0 : 8;
  const __bf16* a0p = A + (size_t)m * lda + klo;
  const __bf16* a1p = A + (size_t)(m + 16) * lda + klo;
  const __bf16* bp  = Brow + klo;
  for (int k0 = 0; k0 < K; k0 += 32) {
    ABu a0, a1, b;
    a0.h[0] = *(const v8bf*)(a0p + k0);
    a0.h[1] = *(const v8bf*)(a0p + k0 + 16);
    a1.h[0] = *(const v8bf*)(a1p + k0);
    a1.h[1] = *(const v8bf*)(a1p + k0 + 16);
    b.h[0]  = *(const v8bf*)(bp + k0);
    b.h[1]  = *(const v8bf*)(bp + k0 + 16);
    __builtin_prefetch(bp + k0 + 256, 0, 1);
    acc0 = __builtin_amdgcn_wmma_f32_16x16x32_bf16(false, a0.v, false, b.v,
                                                   (short)0, acc0, false, false);
    acc1 = __builtin_amdgcn_wmma_f32_16x16x32_bf16(false, a1.v, false, b.v,
                                                   (short)0, acc1, false, false);
  }
}

// ---------------------------------------------------------------------------
// One fused LSTM layer step. Grid = 64 WGs (16 hidden cols each), block = 128
// (4 wave32s). Wave w computes gate w's (i,f,g,o) 32x16 gate tile; gates are
// exchanged through LDS; pointwise sigmoid/tanh cell update in fp32.
// K_IN is a template constant (512 / 1024) so stage counts fold.
// ---------------------------------------------------------------------------
template <int K_IN>
__global__ __launch_bounds__(128) void lstm_step_kernel(
    const __bf16* __restrict__ xin, int x_lda,            // layer input [32,K_IN]
    const __bf16* __restrict__ W_ih,                      // [4H, K_IN] bf16
    const __bf16* __restrict__ W_hh,                      // [4H, H] bf16
    const __bf16* __restrict__ h_prev,                    // [32, H] bf16 (read-only)
    const float*  __restrict__ bias,                      // [4H] = b_ih + b_hh
    float*        __restrict__ c_state,                   // [32, H] fp32, in place
    __bf16*       __restrict__ h_next)                    // [32, H] bf16 (!= h_prev)
{
  __shared__ __align__(16) __bf16 shA[2][32][CHUNK_ + APAD_];  // ~33KB DMA stage
  __shared__ float g_lds[4][32][16];                           // gate exchange (8KB)

  const int tid  = threadIdx.x;
  const int lane = tid & 31;
  const int gate = tid >> 5;            // wave id == gate index (i,f,g,o)
  const int j0   = blockIdx.x * 16;     // hidden-column block
  const int n    = lane & 15;
  const int ncol = gate * HID_ + j0 + n;

  const float bv = bias[ncol];
  v8f acc0 = {bv, bv, bv, bv, bv, bv, bv, bv};
  v8f acc1 = acc0;

  wave_gemm_lds(acc0, acc1, xin,    x_lda, W_ih + (size_t)ncol * K_IN, K_IN,
                lane, tid, shA);
  wave_gemm_lds(acc0, acc1, h_prev, HID_,  W_hh + (size_t)ncol * HID_, HID_,
                lane, tid, shA);

  // C/D layout: VGPR r, lanes 0-15 -> M=r, lanes 16-31 -> M=r+8; N = lane&15.
  const int mb = (lane < 16) ? 0 : 8;
#pragma unroll
  for (int r = 0; r < 8; ++r) {
    g_lds[gate][mb + r][n]      = acc0[r];
    g_lds[gate][mb + r + 16][n] = acc1[r];
  }
  __syncthreads();

#pragma unroll
  for (int q = 0; q < 4; ++q) {
    const int e  = tid * 4 + q;         // 512 (m,jj) elements over 128 threads
    const int m  = e >> 4;
    const int jj = e & 15;
    const float gi = g_lds[0][m][jj];
    const float gf = g_lds[1][m][jj];
    const float gg = g_lds[2][m][jj];
    const float go = g_lds[3][m][jj];
    const size_t idx = (size_t)m * HID_ + j0 + jj;
    const float c  = c_state[idx];
    const float cn = sigmoidf_(gf) * c + sigmoidf_(gi) * tanhf(gg);
    c_state[idx]   = cn;
    h_next[idx]    = to_bf16(sigmoidf_(go) * tanhf(cn));
  }
}

// out[32,512] = h1[32,1024] @ W_out^T + b_out. Grid = 8 WGs x 4 waves.
__global__ __launch_bounds__(128) void readout_kernel(
    const __bf16* __restrict__ h1,
    const __bf16* __restrict__ W_out,   // [OUT, HID] bf16
    const float*  __restrict__ b_out,
    float*        __restrict__ out)
{
  const int tid  = threadIdx.x;
  const int lane = tid & 31;
  const int wv   = tid >> 5;
  const int n    = blockIdx.x * 64 + wv * 16 + (lane & 15);
  const float bv = b_out[n];
  v8f acc0 = {bv, bv, bv, bv, bv, bv, bv, bv};
  v8f acc1 = acc0;
  wave_gemm(acc0, acc1, h1, HID_, W_out + (size_t)n * HID_, HID_, lane);
  const int mb = (lane < 16) ? 0 : 8;
#pragma unroll
  for (int r = 0; r < 8; ++r) {
    out[(size_t)(mb + r) * OUT_ + n]      = acc0[r];
    out[(size_t)(mb + r + 16) * OUT_ + n] = acc1[r];
  }
}

// ---- setup kernels ----
__global__ void f32_to_bf16_kernel(const float* __restrict__ src,
                                   __bf16* __restrict__ dst, int n4) {
  const int i = blockIdx.x * blockDim.x + threadIdx.x;
  if (i < n4) {
    const float4 f = ((const float4*)src)[i];
    union { __bf16 b[4]; uint64_t u; } pk;
    pk.b[0] = to_bf16(f.x); pk.b[1] = to_bf16(f.y);
    pk.b[2] = to_bf16(f.z); pk.b[3] = to_bf16(f.w);
    ((uint64_t*)dst)[i] = pk.u;
  }
}

__global__ void bias_sum_kernel(const float* __restrict__ a,
                                const float* __restrict__ b,
                                float* __restrict__ o, int n) {
  const int i = blockIdx.x * blockDim.x + threadIdx.x;
  if (i < n) o[i] = a[i] + b[i];
}

__global__ void zero_u32_kernel(uint32_t* __restrict__ p, int n) {
  const int i = blockIdx.x * blockDim.x + threadIdx.x;
  if (i < n) p[i] = 0u;
}

extern "C" void kernel_launch(void* const* d_in, const int* in_sizes, int n_in,
                              void* d_out, int out_size, void* d_ws, size_t ws_size,
                              hipStream_t stream) {
  const float* x     = (const float*)d_in[0];
  const float* W_ih0 = (const float*)d_in[1];
  const float* W_hh0 = (const float*)d_in[2];
  const float* b_ih0 = (const float*)d_in[3];
  const float* b_hh0 = (const float*)d_in[4];
  const float* W_ih1 = (const float*)d_in[5];
  const float* W_hh1 = (const float*)d_in[6];
  const float* b_ih1 = (const float*)d_in[7];
  const float* b_hh1 = (const float*)d_in[8];
  const float* W_out = (const float*)d_in[9];
  const float* b_out = (const float*)d_in[10];

  // Workspace bump allocator (~40 MB; bf16 weights stay resident in 192MB L2).
  char* p = (char*)d_ws;
  auto alloc_bf = [&](size_t elems) { __bf16* r = (__bf16*)p; p += elems * sizeof(__bf16); return r; };
  auto alloc_f  = [&](size_t elems) { float*  r = (float*)p;  p += elems * sizeof(float);  return r; };

  __bf16* xb    = alloc_bf((size_t)B_ * T_ * IN_);
  __bf16* Wih0b = alloc_bf((size_t)G4_ * IN_);
  __bf16* Whh0b = alloc_bf((size_t)G4_ * HID_);
  __bf16* Wih1b = alloc_bf((size_t)G4_ * HID_);
  __bf16* Whh1b = alloc_bf((size_t)G4_ * HID_);
  __bf16* Woutb = alloc_bf((size_t)OUT_ * HID_);
  float*  bias0 = alloc_f(G4_);
  float*  bias1 = alloc_f(G4_);
  // state block (contiguous -> single zero-init kernel each call)
  uint32_t* state_base = (uint32_t*)p;
  float*  c0    = alloc_f((size_t)B_ * HID_);
  float*  c1    = alloc_f((size_t)B_ * HID_);
  __bf16* h0[2] = { alloc_bf((size_t)B_ * HID_), alloc_bf((size_t)B_ * HID_) };
  __bf16* h1[2] = { alloc_bf((size_t)B_ * HID_), alloc_bf((size_t)B_ * HID_) };
  const int state_words = (int)(((char*)p - (char*)state_base) / 4);

  auto conv = [&](const float* s, __bf16* dst, size_t elems) {
    const int n4 = (int)(elems / 4);
    f32_to_bf16_kernel<<<(n4 + 255) / 256, 256, 0, stream>>>(s, dst, n4);
  };
  conv(x,     xb,    (size_t)B_ * T_ * IN_);
  conv(W_ih0, Wih0b, (size_t)G4_ * IN_);
  conv(W_hh0, Whh0b, (size_t)G4_ * HID_);
  conv(W_ih1, Wih1b, (size_t)G4_ * HID_);
  conv(W_hh1, Whh1b, (size_t)G4_ * HID_);
  conv(W_out, Woutb, (size_t)OUT_ * HID_);
  bias_sum_kernel<<<(G4_ + 255) / 256, 256, 0, stream>>>(b_ih0, b_hh0, bias0, G4_);
  bias_sum_kernel<<<(G4_ + 255) / 256, 256, 0, stream>>>(b_ih1, b_hh1, bias1, G4_);
  zero_u32_kernel<<<(state_words + 255) / 256, 256, 0, stream>>>(state_base, state_words);

  // Sequential scan: one fused kernel per layer per timestep (graph-captured).
  for (int t = 0; t < T_; ++t) {
    const int w = t & 1, r = w ^ 1;
    // layer 0: input = x[:, t, :], row stride T*IN
    lstm_step_kernel<IN_><<<HID_ / 16, 128, 0, stream>>>(
        xb + (size_t)t * IN_, T_ * IN_,
        Wih0b, Whh0b, h0[r], bias0, c0, h0[w]);
    // layer 1: input = h0(t)
    lstm_step_kernel<HID_><<<HID_ / 16, 128, 0, stream>>>(
        h0[w], HID_,
        Wih1b, Whh1b, h1[r], bias1, c1, h1[w]);
  }

  readout_kernel<<<OUT_ / 64, 128, 0, stream>>>(
      h1[(T_ - 1) & 1], Woutb, b_out, (float*)d_out);
}
